// Attention_73126113182349
// MI455X (gfx1250) — matrix operand order repaired and checked by reading, not compiled, not run on hardware
//
#include <hip/hip_runtime.h>
#include <stdint.h>

// ---------------------------------------------------------------------------
// Attention layer for MI455X (gfx1250, wave32, WMMA + TDM).
//   B=4, S=2048, H=1024, NH=16, HD=64.
// Compute-bound (~137 GFLOP vs ~0.15 GB HBM): all matmuls via
// v_wmma_f32_16x16x32_bf16 (fp32 accum). Flash attention keeps 2048^2 scores
// in VGPRs. bf16 tiles (attention K, ctx A-tiles) are DMA'd HBM->LDS with
// tensor_load_to_lds (TENSORcnt), ping-pong double buffered against WMMA.
// ---------------------------------------------------------------------------

typedef __attribute__((ext_vector_type(16))) __bf16        v16bf;
typedef __attribute__((ext_vector_type(8)))  float         v8f;
typedef __attribute__((ext_vector_type(4)))  unsigned int  v4u;
typedef __attribute__((ext_vector_type(8)))  unsigned int  v8u;

static constexpr int kB  = 4;
static constexpr int kS  = 2048;
static constexpr int kH  = 1024;
static constexpr int kNH = 16;
static constexpr int kHD = 64;
static constexpr int kM  = kB * kS;        // 8192 GEMM rows
static constexpr int kPad = 40;            // padded tile row stride (u16): 80B,
                                           // 16B aligned, 20 banks -> conflict free

union FragU { uint4 u[2]; v16bf v; };

__device__ __forceinline__ unsigned short f2bf(float f) {
  union { float f; unsigned int u; } c; c.f = f;
  unsigned int u = c.u;
  return (unsigned short)((u + 0x7FFFu + ((u >> 16) & 1u)) >> 16);  // RNE
}

// A fragment (16 x K=32 bf16): lane m=l&15; VGPR0-3 K=(l>>4)*8..+7,
// VGPR4-7 K=16+(l>>4)*8..+7. memcpy => ds/global _load_b128, safe aliasing.
__device__ __forceinline__ v16bf frag_a(const unsigned short* base, int tile_row,
                                        int stride_u16, int lane) {
  int off = (tile_row + (lane & 15)) * stride_u16 + ((lane >> 4) << 3);
  FragU f;
  __builtin_memcpy(&f.u[0], base + off,      16);
  __builtin_memcpy(&f.u[1], base + off + 16, 16);
  return f.v;
}
// B fragment (K=32 x 16 bf16): lane n=l&15; VGPR0-3 K=(l>>4)*8..+7,
// VGPR4-7 K=16+(l>>4)*8..+7 (same K order as frag_a).
__device__ __forceinline__ v16bf frag_b(const unsigned short* base, int tile_col,
                                        int stride_u16, int lane) {
  int off = (tile_col + (lane & 15)) * stride_u16 + ((lane >> 4) << 3);
  FragU f;
  __builtin_memcpy(&f.u[0], base + off,      16);
  __builtin_memcpy(&f.u[1], base + off + 16, 16);
  return f.v;
}

__device__ __forceinline__ float rmax16(float v) {
  #pragma unroll
  for (int m = 1; m < 16; m <<= 1) v = fmaxf(v, __shfl_xor(v, m, 16));
  return v;
}
__device__ __forceinline__ float rsum16(float v) {
  #pragma unroll
  for (int m = 1; m < 16; m <<= 1) v += __shfl_xor(v, m, 16);
  return v;
}

// ---------------------------------------------------------------------------
// Tensor Data Mover: async 2-D bf16 tile HBM -> LDS (CDNA5 ISA, D# groups
// 0/1; VADDR2/3 NULL for 2-D). Issue from one wave; sync: s_wait_tensorcnt + bar.
//   flags: group1 dword0 (data_size=2B always; optional LDS pad fields).
// ---------------------------------------------------------------------------
__device__ __forceinline__ void tdm_load_2d(unsigned int lds_byte_addr,
                                            const void* gsrc,
                                            unsigned int dim0, unsigned int dim1,
                                            unsigned int tile0, unsigned int tile1,
                                            unsigned int stride0,
                                            unsigned int flags) {
  uint64_t ga = (uint64_t)(uintptr_t)gsrc;
  v4u g0 = { 1u,                                  // count=1 (valid user D#)
             lds_byte_addr,                       // lds_addr
             (unsigned int)ga,                    // global_addr[31:0]
             (unsigned int)((ga >> 32) & 0x01FFFFFFu) | 0x80000000u }; // +type=2
  v8u g1 = { flags,                               // wg_mask=0|data_size=1(2B)|pad
             (dim0 & 0xFFFFu) << 16,              // tensor_dim0[15:0]
             (dim0 >> 16) | ((dim1 & 0xFFFFu) << 16),
             (dim1 >> 16) | (tile0 << 16),        // tile_dim0
             tile1,                               // tile_dim1 (tile_dim2=0)
             stride0, 0u, 0u };                   // tensor_dim0_stride[47:0]
  asm volatile("tensor_load_to_lds %0, %1" :: "s"(g0), "s"(g1) : "memory");
}
__device__ __forceinline__ unsigned int lds_addr_of(const void* p) {
  return (unsigned int)(uintptr_t)p;              // flat LDS aperture: addr[31:0]
}
static constexpr unsigned kTdmPlain = 0x00010000u;             // data_size=2B
// pad_enable | pad_interval=16dw (64B) | pad_amount=4dw (16B): 64B rows -> 80B
static constexpr unsigned kTdmPad80 = 0x00010000u | (1u << 20) | (3u << 22) | (3u << 25);

// ---------------------------------------------------------------------------
// GEMM: C[8192x1024] = A * W + bias. Block tile 128x256, 8 waves of 64x64
// (16 accumulators), K-step 32, ping-pong LDS (1 barrier / step).
// modes 0/1/2: A = fp32 hidden -> bf16; out bf16 ws [b,nh,s,hd] (+fp32 K/V).
// mode 3:      A = bf16 ctx, staged by TDM (pad->80B rows); out fp32 row-major.
// ---------------------------------------------------------------------------
__global__ __launch_bounds__(256) void gemm_wmma(
    const float* __restrict__ X, const unsigned short* __restrict__ Xbf,
    const float* __restrict__ Wq, const float* __restrict__ Wk,
    const float* __restrict__ Wv, const float* __restrict__ Wp,
    const float* __restrict__ bq, const float* __restrict__ bk,
    const float* __restrict__ bv, const float* __restrict__ bp,
    unsigned short* __restrict__ q_ws, unsigned short* __restrict__ k_ws,
    unsigned short* __restrict__ v_ws,
    float* __restrict__ out32, float* __restrict__ k32, float* __restrict__ v32,
    int mode_base) {
  const int mode = mode_base + (int)blockIdx.z;
  const float* W;  const float* bias;
  unsigned short* obf; float* o32;
  switch (mode) {
    case 0:  W = Wq; bias = bq; obf = q_ws;    o32 = nullptr; break;
    case 1:  W = Wk; bias = bk; obf = k_ws;    o32 = k32;     break;
    case 2:  W = Wv; bias = bv; obf = v_ws;    o32 = v32;     break;
    default: W = Wp; bias = bp; obf = nullptr; o32 = out32;   break;
  }

  __shared__ unsigned short As[2][128 * kPad];   // [m][k] bf16, 2 x 10 KB
  __shared__ unsigned short Bs[2][256 * kPad];   // [n][k] bf16 (W^T), 2 x 20 KB

  const int tid = threadIdx.x, lane = tid & 31, wid = tid >> 5;
  const int wM = wid & 1, wN = wid >> 1;              // 2 x 4 wave grid
  const int mbase = blockIdx.x * 128, nbase = blockIdx.y * 256;
  constexpr int KT = kH / 32;

  const v8f zero8 = {0.f, 0.f, 0.f, 0.f, 0.f, 0.f, 0.f, 0.f};
  v8f acc[4][4];
  #pragma unroll
  for (int mi = 0; mi < 4; ++mi)
    #pragma unroll
    for (int ni = 0; ni < 4; ++ni) acc[mi][ni] = zero8;

  // ---- tile staging (one K-step) into buffer `bi` ----
  auto stage = [&](int kt, int bi) {
    const int kb = kt * 32;
    if (mode < 3) {                       // A: fp32 -> bf16 manual
      #pragma unroll
      for (int i = 0; i < 4; ++i) {
        int idx = tid + i * 256;          // 1024 float4 chunks
        int row = idx >> 3, c4 = idx & 7;
        float4 xv;
        __builtin_memcpy(&xv, &X[(size_t)(mbase + row) * kH + kb + c4 * 4], 16);
        uint2 p;
        p.x = (unsigned)f2bf(xv.x) | ((unsigned)f2bf(xv.y) << 16);
        p.y = (unsigned)f2bf(xv.z) | ((unsigned)f2bf(xv.w) << 16);
        *(uint2*)&As[bi][row * kPad + c4 * 4] = p;
      }
    } else if (wid == 0) {                // A: bf16 ctx via TDM, rows pad to 80B
      tdm_load_2d(lds_addr_of(&As[bi][0]),
                  &Xbf[(size_t)mbase * kH + kb],
                  kH, kM, 32, 128, kH, kTdmPad80);
    }
    #pragma unroll
    for (int i = 0; i < 8; ++i) {         // B: W[kb..+31][nbase..+255] -> [n][k]
      int idx = tid + i * 256;            // 2048 float4 chunks
      int kk = idx >> 6, n4 = idx & 63;
      float4 wv;
      __builtin_memcpy(&wv, &W[(size_t)(kb + kk) * kH + nbase + n4 * 4], 16);
      Bs[bi][(n4 * 4 + 0) * kPad + kk] = f2bf(wv.x);
      Bs[bi][(n4 * 4 + 1) * kPad + kk] = f2bf(wv.y);
      Bs[bi][(n4 * 4 + 2) * kPad + kk] = f2bf(wv.z);
      Bs[bi][(n4 * 4 + 3) * kPad + kk] = f2bf(wv.w);
    }
  };

  stage(0, 0);
  for (int kt = 0; kt < KT; ++kt) {
    __builtin_amdgcn_s_wait_tensorcnt(0);          // TDM tile (mode 3) landed
    __syncthreads();                               // tile kt visible; prev reads done
    if (kt + 1 < KT) stage(kt + 1, (kt + 1) & 1);  // overlap with WMMA below
    const unsigned short* Ab = As[kt & 1];
    const unsigned short* Bb = Bs[kt & 1];
    v16bf af[4];
    #pragma unroll
    for (int mi = 0; mi < 4; ++mi) af[mi] = frag_a(Ab, wM * 64 + mi * 16, kPad, lane);
    #pragma unroll
    for (int ni = 0; ni < 4; ++ni) {
      v16bf bfr = frag_b(Bb, wN * 64 + ni * 16, kPad, lane);
      #pragma unroll
      for (int mi = 0; mi < 4; ++mi)
        acc[mi][ni] = __builtin_amdgcn_wmma_f32_16x16x32_bf16(
            false, af[mi], false, bfr, (short)0, acc[mi][ni], false, false);
    }
  }

  // ---- epilogue: bias + scatter ----
  #pragma unroll
  for (int mi = 0; mi < 4; ++mi)
    #pragma unroll
    for (int ni = 0; ni < 4; ++ni)
      #pragma unroll
      for (int r = 0; r < 8; ++r) {
        int row = mbase + wM * 64 + mi * 16 + r + ((lane >> 4) << 3);
        int col = nbase + wN * 64 + ni * 16 + (lane & 15);
        float v = acc[mi][ni][r] + bias[col];
        if (mode == 3) {
          o32[(size_t)row * kH + col] = v;
        } else {
          int b = row >> 11, s = row & (kS - 1);
          int h = col >> 6,  d = col & (kHD - 1);
          size_t idx = (((size_t)(b * kNH + h)) * kS + s) * kHD + d;
          obf[idx] = f2bf(v);
          if (o32) o32[idx] = v;
        }
      }
}

// ---------------------------------------------------------------------------
// Causal flash attention. Block = (bh, 128 q-rows), 8 waves x 16 rows.
// K tiles: TDM HBM->LDS (async, TENSORcnt). V tiles: manual transpose stage.
// Ping-pong double buffer, 1 barrier per 32-key step.
// ---------------------------------------------------------------------------
__global__ __launch_bounds__(256) void attn_flash(
    const unsigned short* __restrict__ q_ws,
    const unsigned short* __restrict__ k_ws,
    const unsigned short* __restrict__ v_ws,
    unsigned short* __restrict__ ctx_ws) {
  __shared__ unsigned short Ks[2][32 * 64];        // [key][hd], TDM target, 2x4 KB
  __shared__ unsigned short Vt[2][64 * kPad];      // [hd][key] padded, 2x5 KB
  __shared__ unsigned short Pb[8 * 16 * kPad];     // per-wave P staging, 10 KB

  const int tid = threadIdx.x, lane = tid & 31, wid = tid >> 5;
  const int bh = blockIdx.y, b = bh >> 4, head = bh & 15;
  const int qbase = blockIdx.x * 128;
  const int q0 = qbase + wid * 16;
  const int col = lane & 15, hs = lane >> 4;

  const unsigned short* qp = q_ws + (size_t)bh * kS * kHD;
  const unsigned short* kp = k_ws + (size_t)bh * kS * kHD;
  const unsigned short* vp = v_ws + (size_t)bh * kS * kHD;
  unsigned short* Pw = Pb + wid * 16 * kPad;

  v16bf aQ[2];
  #pragma unroll
  for (int s = 0; s < 2; ++s) aQ[s] = frag_a(qp + s * 32, q0, kHD, lane);

  const v8f zero8 = {0.f, 0.f, 0.f, 0.f, 0.f, 0.f, 0.f, 0.f};
  v8f O[4]; float mrow[8], lrow[8];
  #pragma unroll
  for (int t = 0; t < 4; ++t) O[t] = zero8;
  #pragma unroll
  for (int r = 0; r < 8; ++r) { mrow[r] = -1e30f; lrow[r] = 0.f; }

  const int nkt = (qbase >> 5) + 4;                // causal: keys <= qbase+127

  auto stage = [&](int kt, int bi) {
    const int k0 = kt * 32;
    if (wid == 0)                                  // K tile: async DMA, no VGPRs
      tdm_load_2d(lds_addr_of(&Ks[bi][0]), kp + (size_t)k0 * kHD,
                  kHD, kS, kHD, 32, kHD, kTdmPlain);
    int row = tid >> 3, c8 = tid & 7;              // V tile: load + transpose
    uint4 vv;
    __builtin_memcpy(&vv, &vp[(size_t)(k0 + row) * kHD + c8 * 8], 16);
    const unsigned short* pv = (const unsigned short*)&vv;
    #pragma unroll
    for (int j = 0; j < 8; ++j) Vt[bi][(c8 * 8 + j) * kPad + row] = pv[j];
  };

  stage(0, 0);
  for (int kt = 0; kt < nkt; ++kt) {
    __builtin_amdgcn_s_wait_tensorcnt(0);          // wave0's K-tile DMA done
    __syncthreads();                               // tile kt visible to block
    if (kt + 1 < nkt) stage(kt + 1, (kt + 1) & 1);
    const int k0 = kt * 32;
    const unsigned short* Kb = Ks[kt & 1];
    const unsigned short* Vb = Vt[kt & 1];

    if (k0 <= q0 + 15) {
      v8f S0 = zero8, S1 = zero8;
      #pragma unroll
      for (int s = 0; s < 2; ++s) {                // contract hd in 32-slices
        v16bf b0 = frag_b(Kb + s * 32, 0,  kHD, lane);
        v16bf b1 = frag_b(Kb + s * 32, 16, kHD, lane);
        S0 = __builtin_amdgcn_wmma_f32_16x16x32_bf16(false, aQ[s], false, b0,
                                                     (short)0, S0, false, false);
        S1 = __builtin_amdgcn_wmma_f32_16x16x32_bf16(false, aQ[s], false, b1,
                                                     (short)0, S1, false, false);
      }
      #pragma unroll
      for (int r = 0; r < 8; ++r) {                // online softmax per row
        int qr = q0 + r + (hs << 3);
        float a0 = (k0 + col      <= qr) ? S0[r] * 0.125f : -1e30f;
        float a1 = (k0 + 16 + col <= qr) ? S1[r] * 0.125f : -1e30f;
        float mnew  = fmaxf(mrow[r], rmax16(fmaxf(a0, a1)));
        float alpha = __expf(mrow[r] - mnew);
        float p0 = __expf(a0 - mnew), p1 = __expf(a1 - mnew);
        lrow[r] = lrow[r] * alpha + rsum16(p0 + p1);
        mrow[r] = mnew;
        #pragma unroll
        for (int t = 0; t < 4; ++t) O[t][r] *= alpha;
        int prow = r + (hs << 3);
        Pw[prow * kPad + col]      = f2bf(p0);     // C-layout -> A staging
        Pw[prow * kPad + 16 + col] = f2bf(p1);
      }
      v16bf aP = frag_a(Pw, 0, kPad, lane);        // same-wave DS: in order
      #pragma unroll
      for (int t = 0; t < 4; ++t) {
        v16bf bV = frag_b(Vb, t * 16, kPad, lane);
        O[t] = __builtin_amdgcn_wmma_f32_16x16x32_bf16(false, aP, false, bV,
                                                       (short)0, O[t], false, false);
      }
    }
  }

  #pragma unroll
  for (int t = 0; t < 4; ++t)
    #pragma unroll
    for (int r = 0; r < 8; ++r) {
      int row = q0 + r + (hs << 3);
      int hc = head * kHD + t * 16 + col;
      ctx_ws[((size_t)(b * kS + row)) * kH + hc] = f2bf(O[t][r] / lrow[r]);
    }
}

// ---------------------------------------------------------------------------
extern "C" void kernel_launch(void* const* d_in, const int* in_sizes, int n_in,
                              void* d_out, int out_size, void* d_ws, size_t ws_size,
                              hipStream_t stream) {
  (void)in_sizes; (void)n_in; (void)out_size; (void)ws_size;
  const float* hidden = (const float*)d_in[0];
  const float* Wq = (const float*)d_in[1]; const float* bq = (const float*)d_in[2];
  const float* Wk = (const float*)d_in[3]; const float* bk = (const float*)d_in[4];
  const float* Wv = (const float*)d_in[5]; const float* bv = (const float*)d_in[6];
  const float* Wp = (const float*)d_in[7]; const float* bp = (const float*)d_in[8];

  const size_t per = (size_t)kB * kNH * kS * kHD;     // 8,388,608 elems
  float* out  = (float*)d_out;
  float* kout = out + per;
  float* vout = out + 2 * per;

  unsigned short* q_ws   = (unsigned short*)d_ws;     // 4 x 16.8 MB bf16 in ws
  unsigned short* k_ws   = q_ws + per;
  unsigned short* v_ws   = q_ws + 2 * per;
  unsigned short* ctx_ws = q_ws + 3 * per;

  dim3 blk(256);
  // QKV projections (z = Q,K,V); K/V also stored fp32 to d_out.
  gemm_wmma<<<dim3(kM / 128, kH / 256, 3), blk, 0, stream>>>(
      hidden, ctx_ws, Wq, Wk, Wv, Wp, bq, bk, bv, bp,
      q_ws, k_ws, v_ws, out, kout, vout, 0);
  // Causal flash attention -> ctx (bf16 ws).
  attn_flash<<<dim3(kS / 128, kB * kNH), blk, 0, stream>>>(q_ws, k_ws, v_ws, ctx_ws);
  // Output projection: ctx @ Wp + bp -> out (fp32), A-tiles via TDM.
  gemm_wmma<<<dim3(kM / 128, kH / 256, 1), blk, 0, stream>>>(
      hidden, ctx_ws, Wq, Wk, Wv, Wp, bq, bk, bv, bp,
      q_ws, k_ws, v_ws, out, kout, vout, 3);
}
